// LstmCrfModel_52974126629062
// MI455X (gfx1250) — matrix operand order repaired and checked by reading, not compiled, not run on hardware
//
#include <hip/hip_runtime.h>

// ---------------------------------------------------------------------------
// Model dims
#define Vv 100000
#define Ee 128
#define Hh 512
#define Ll 9
#define Bb 64
#define Tt 512
#define G4 2048            // 4*H
#define Mrows (Bb*Tt)      // 32768
#define HROW (Tt+1)        // padded time dim for h history (t=0 row is zeros)
#define SROW (Hh+8)        // LDS row stride (bf16 elems), 16B pad kills bank conflicts

// Workspace layout (bytes)
#define OFF_X    0ull                         // x gathered, bf16  (32768*128*2 = 8 MB)
#define OFF_WXT  8388608ull                   // Wx^T bf16 (2048*128*2)
#define OFF_WHT  8912896ull                   // Wh^T bf16 (2048*512*2)
#define OFF_WDT  11010048ull                  // Wd^T padded bf16 (16*512*2)
#define OFF_C    11026432ull                  // c state fp32 (64*512*4)
#define OFF_HALL 11681792ull                  // h history bf16 (64*513*512*2)
#define OFF_XG   45301760ull                  // xg fp32 (32768*2048*4 = 268 MB)

typedef __attribute__((ext_vector_type(16))) __bf16 v16bf;
typedef __attribute__((ext_vector_type(8)))  __bf16 v8bf;
typedef __attribute__((ext_vector_type(8)))  float  v8f;

union BF16x16 { v16bf v; v8bf h[2]; };

// A fragment: 16x32 bf16, row-major source, row stride ldk (elements).
// ISA 7.12.2 16-bit A 16x32: lanes 0-15 -> K {0..7,16..23}, lanes 16-31 ->
// K {8..15,24..31}; row M = lane&15.
__device__ __forceinline__ v16bf load_a_frag(const __bf16* base, int ldk, int lane) {
  int row  = lane & 15;
  int half = lane >> 4;
  const __bf16* p = base + (long)row * ldk + half * 8;
  BF16x16 r;
  r.h[0] = *(const v8bf*)(p);
  r.h[1] = *(const v8bf*)(p + 16);
  return r.v;
}

// B fragment: 32x16 bf16 from an N-major (transposed) weight matrix.
__device__ __forceinline__ v16bf load_b_frag(const __bf16* wt, int ldk, int lane) {
  int n    = lane & 15;
  int half = lane >> 4;
  const __bf16* p = wt + (long)n * ldk + half * 8;
  BF16x16 r;
  r.h[0] = *(const v8bf*)(p);
  r.h[1] = *(const v8bf*)(p + 16);
  return r.v;
}

__device__ __forceinline__ float sigmoidf_(float x) {
  return 1.0f / (1.0f + __expf(-x));
}

// ---------------------------------------------------------------------------
// Prep kernels
__global__ void convT_kernel(const float* __restrict__ in, __bf16* __restrict__ out,
                             int K, int N) {              // out[N][K] = in[K][N]
  int idx = blockIdx.x * 256 + threadIdx.x;
  if (idx >= N * K) return;
  int n = idx / K, k = idx - n * K;
  out[idx] = (__bf16)in[k * N + n];
}

__global__ void convT_pad_kernel(const float* __restrict__ Wd, __bf16* __restrict__ out) {
  int idx = blockIdx.x * 256 + threadIdx.x;               // out 16 x 512
  if (idx >= 16 * Hh) return;
  int n = idx / Hh, k = idx - n * Hh;
  out[idx] = (n < Ll) ? (__bf16)Wd[k * Ll + n] : (__bf16)0.0f;
}

__global__ void init_state_kernel(float* __restrict__ C, __bf16* __restrict__ Hall) {
  int idx = blockIdx.x * 256 + threadIdx.x;               // 64*512
  C[idx] = 0.0f;
  int b = idx >> 9, j = idx & 511;
  Hall[((long)b * HROW) * Hh + j] = (__bf16)0.0f;         // zero t=0 row
}

__global__ void gather_kernel(const int* __restrict__ text, const float* __restrict__ embed,
                              __bf16* __restrict__ X) {
  int idx = blockIdx.x * 256 + threadIdx.x;               // 32768*128
  int m = idx >> 7, e = idx & 127;
  X[idx] = (__bf16)embed[(long)text[m] * Ee + e];
}

// ---------------------------------------------------------------------------
// xg = x @ Wx + b      M=32768  K=128  N=2048   (wave computes 16x64 tile)
__global__ void xg_gemm_kernel(const __bf16* __restrict__ X, const __bf16* __restrict__ WxT,
                               const float* __restrict__ bias, float* __restrict__ XG) {
  int wave = (blockIdx.x * blockDim.x + threadIdx.x) >> 5;
  int lane = threadIdx.x & 31;
  int mt = wave >> 5;          // 2048 row tiles
  int nt = wave & 31;          // 32 col tiles of 64
  int m0 = mt * 16, n0 = nt * 64;
  const __bf16* a_base = X + (long)m0 * Ee;
  v8f acc[4] = {v8f{}, v8f{}, v8f{}, v8f{}};
#pragma unroll
  for (int kb = 0; kb < Ee; kb += 32) {
    v16bf a = load_a_frag(a_base + kb, Ee, lane);
#pragma unroll
    for (int j = 0; j < 4; ++j) {
      v16bf bf = load_b_frag(WxT + (long)(n0 + j * 16) * Ee + kb, Ee, lane);
      acc[j] = __builtin_amdgcn_wmma_f32_16x16x32_bf16(false, a, false, bf,
                                                       (short)0, acc[j], false, false);
    }
  }
  int half = lane >> 4, nl = lane & 15;
#pragma unroll
  for (int j = 0; j < 4; ++j) {
    int n = n0 + j * 16 + nl;
    float bv = bias[n];
#pragma unroll
    for (int r = 0; r < 8; ++r) {
      int m = m0 + r + half * 8;
      XG[(long)m * G4 + n] = acc[j][r] + bv;
    }
  }
}

// ---------------------------------------------------------------------------
// Fused LSTM step: one wave computes the 16x16 (batch, hidden) tile for ALL
// FOUR gate blocks (identical C layout), then applies the nonlinearity and
// state update in-register. h m-tile (16x512 bf16) staged to LDS once per WG;
// A-fragments then come from ds_load_b128.
// Grid: 16 blocks x 256 threads = 128 waves = 4 m-tiles x 32 j-tiles.
__global__ void lstm_step_kernel(__bf16* __restrict__ Hall,
                                 const __bf16* __restrict__ WhT,
                                 const float* __restrict__ XG,
                                 float* __restrict__ C, int t) {
  __shared__ __bf16 sA[16 * SROW];
  int wave = threadIdx.x >> 5;
  int lane = threadIdx.x & 31;
  int mt = blockIdx.x >> 2;                  // 4 m-tiles (batch groups of 16)
  int jt = (blockIdx.x & 3) * 8 + wave;      // 32 hidden tiles of 16
  int m0 = mt * 16, j0 = jt * 16;

  // Prefetch this wave's Wh columns (global_prefetch_b8; L2-resident after step 0)
#pragma unroll
  for (int g = 0; g < 4; ++g)
    __builtin_prefetch(WhT + (long)(g * Hh + j0 + (lane >> 1)) * Hh + (lane & 1) * 256, 0, 1);

  // Stage h_{t-1}[m0:m0+16, :] into LDS (1024 x 16B chunks, 4 per thread)
  for (int i = threadIdx.x; i < 1024; i += 256) {
    int r = i >> 6, c = (i & 63) * 8;
    const __bf16* gp = Hall + ((long)(m0 + r) * HROW + t) * Hh + c;
    *(v8bf*)(sA + r * SROW + c) = *(const v8bf*)gp;
  }
  __syncthreads();

  v8f acc[4] = {v8f{}, v8f{}, v8f{}, v8f{}};
  for (int kb = 0; kb < Hh; kb += 32) {
    v16bf a = load_a_frag(sA + kb, SROW, lane);   // ds_load_b128 x2
#pragma unroll
    for (int g = 0; g < 4; ++g) {                  // gate blocks i,f,c,o
      v16bf bf = load_b_frag(WhT + (long)(g * Hh + j0) * Hh + kb, Hh, lane);
      acc[g] = __builtin_amdgcn_wmma_f32_16x16x32_bf16(false, a, false, bf,
                                                       (short)0, acc[g], false, false);
    }
  }

  // Gate math: identical C layout across the 4 accumulators -> purely per-lane
  int half = lane >> 4, nl = lane & 15;
  int j = j0 + nl;
#pragma unroll
  for (int r = 0; r < 8; ++r) {
    int b = m0 + r + half * 8;
    const float* xg = XG + ((long)b * Tt + t) * G4;
    float gi = acc[0][r] + xg[j];
    float gf = acc[1][r] + xg[Hh + j];
    float gc = acc[2][r] + xg[2 * Hh + j];
    float go = acc[3][r] + xg[3 * Hh + j];
    long cidx = (long)b * Hh + j;
    float c = sigmoidf_(gf) * C[cidx] + sigmoidf_(gi) * tanhf(gc);
    float h = sigmoidf_(go) * tanhf(c);
    C[cidx] = c;
    Hall[((long)b * HROW + t + 1) * Hh + j] = (__bf16)h;
  }
}

// ---------------------------------------------------------------------------
// logits = h @ Wd + bd    M=32768  K=512  N=9 (padded to 16). Wave: 16x16 tile.
__global__ void logits_gemm_kernel(const __bf16* __restrict__ Hall, const __bf16* __restrict__ WdT,
                                   const float* __restrict__ bd, float* __restrict__ out) {
  int wave = blockIdx.x * 8 + (threadIdx.x >> 5);  // 2048 waves
  int lane = threadIdx.x & 31;
  int m0 = wave * 16;
  int b  = m0 >> 9;           // 512 % 16 == 0 -> a tile never crosses batch rows
  int t0 = m0 & 511;
  const __bf16* a_base = Hall + ((long)b * HROW + t0 + 1) * Hh;  // row stride Hh (contig t)
  v8f acc = v8f{};
  for (int kb = 0; kb < Hh; kb += 32) {
    v16bf a  = load_a_frag(a_base + kb, Hh, lane);
    v16bf bf = load_b_frag(WdT + kb, Hh, lane);
    acc = __builtin_amdgcn_wmma_f32_16x16x32_bf16(false, a, false, bf,
                                                  (short)0, acc, false, false);
  }
  int half = lane >> 4, nl = lane & 15;
  if (nl < Ll) {
    float bv = bd[nl];
#pragma unroll
    for (int r = 0; r < 8; ++r) {
      int m = m0 + r + half * 8;
      out[(long)m * Ll + nl] = acc[r] + bv;
    }
  }
}

// ---------------------------------------------------------------------------
// CRF log-likelihood: one wave (32 lanes) per batch row; lanes 0..8 carry alpha.
__global__ void crf_kernel(const int* __restrict__ text, const int* __restrict__ labels,
                           const float* __restrict__ trans, const float* __restrict__ logits,
                           float* __restrict__ lens_out, float* __restrict__ ll_out) {
  int b = blockIdx.x;
  int lane = threadIdx.x;

  // lens[b] = count(text[b,:] != 0)
  int cnt = 0;
  for (int t = lane; t < Tt; t += 32) cnt += (text[b * Tt + t] != 0) ? 1 : 0;
  for (int o = 16; o > 0; o >>= 1) cnt += __shfl_xor(cnt, o, 32);
  int len = __shfl(cnt, 0, 32);
  if (lane == 0) lens_out[b] = (float)len;

  int j = (lane < Ll) ? lane : 0;        // guard OOB reads on inactive lanes
  float trj[Ll];
#pragma unroll
  for (int i = 0; i < Ll; ++i) trj[i] = trans[i * Ll + j];

  float alpha = logits[(long)(b * Tt) * Ll + j];
  for (int t = 1; t < Tt; ++t) {
    float lg = logits[(long)(b * Tt + t) * Ll + j];
    float v[Ll];
#pragma unroll
    for (int i = 0; i < Ll; ++i) v[i] = __shfl(alpha, i, 32) + trj[i];
    float mx = v[0];
#pragma unroll
    for (int i = 1; i < Ll; ++i) mx = fmaxf(mx, v[i]);
    float s = 0.0f;
#pragma unroll
    for (int i = 0; i < Ll; ++i) s += __expf(v[i] - mx);
    float nw = mx + __logf(s) + lg;
    if (t < len) alpha = nw;
  }

  // log_norm = logsumexp over lanes 0..8
  float m2 = -3.4e38f, s2 = 0.0f;
#pragma unroll
  for (int i = 0; i < Ll; ++i) m2 = fmaxf(m2, __shfl(alpha, i, 32));
#pragma unroll
  for (int i = 0; i < Ll; ++i) s2 += __expf(__shfl(alpha, i, 32) - m2);
  float log_norm = m2 + __logf(s2);

  // score = unary + binary
  float sc = 0.0f;
  for (int t = lane; t < Tt; t += 32) {
    if (t < len) {
      int lab = labels[b * Tt + t];
      sc += logits[(long)(b * Tt + t) * Ll + lab];
      if (t >= 1) sc += trans[labels[b * Tt + t - 1] * Ll + lab];
    }
  }
  for (int o = 16; o > 0; o >>= 1) sc += __shfl_xor(sc, o, 32);
  if (lane == 0) ll_out[b] = sc - log_norm;
}

// ---------------------------------------------------------------------------
extern "C" void kernel_launch(void* const* d_in, const int* in_sizes, int n_in,
                              void* d_out, int out_size, void* d_ws, size_t ws_size,
                              hipStream_t stream) {
  const int*   text   = (const int*)d_in[0];
  const int*   labels = (const int*)d_in[1];
  const float* embed  = (const float*)d_in[2];
  const float* Wx     = (const float*)d_in[3];
  const float* Wh     = (const float*)d_in[4];
  const float* bias   = (const float*)d_in[5];
  const float* Wd     = (const float*)d_in[6];
  const float* bd     = (const float*)d_in[7];
  const float* trans  = (const float*)d_in[8];
  float* out = (float*)d_out;

  char* ws = (char*)d_ws;
  __bf16* X    = (__bf16*)(ws + OFF_X);
  __bf16* WxT  = (__bf16*)(ws + OFF_WXT);
  __bf16* WhT  = (__bf16*)(ws + OFF_WHT);
  __bf16* WdT  = (__bf16*)(ws + OFF_WDT);
  float*  Cst  = (float*) (ws + OFF_C);
  __bf16* Hall = (__bf16*)(ws + OFF_HALL);
  float*  XG   = (float*) (ws + OFF_XG);

  // Weight transposes + state init + embedding gather
  convT_kernel<<<(G4 * Ee + 255) / 256, 256, 0, stream>>>(Wx, WxT, Ee, G4);
  convT_kernel<<<(G4 * Hh + 255) / 256, 256, 0, stream>>>(Wh, WhT, Hh, G4);
  convT_pad_kernel<<<(16 * Hh + 255) / 256, 256, 0, stream>>>(Wd, WdT);
  init_state_kernel<<<(Bb * Hh) / 256, 256, 0, stream>>>(Cst, Hall);
  gather_kernel<<<(Mrows * Ee) / 256, 256, 0, stream>>>(text, embed, X);

  // Input projection (big parallel WMMA GEMM)
  xg_gemm_kernel<<<(Mrows / 16) * 32 / 8, 256, 0, stream>>>(X, WxT, bias, XG);

  // Serial LSTM recurrence: 512 fused (WMMA + gates) steps, one kernel each
  for (int t = 0; t < Tt; ++t)
    lstm_step_kernel<<<16, 256, 0, stream>>>(Hall, WhT, XG, Cst, t);

  // Decoder GEMM -> logits straight into d_out
  logits_gemm_kernel<<<(Mrows / 16) / 8, 256, 0, stream>>>(Hall, WdT, bd, out);

  // CRF: lens + log-likelihood (reads logits from d_out)
  crf_kernel<<<Bb, 32, 0, stream>>>(text, labels, trans, out,
                                    out + (long)Mrows * Ll,
                                    out + (long)Mrows * Ll + Bb);
}
